// HSIC_71236327572062
// MI455X (gfx1250) — matrix operand order repaired and checked by reading, not compiled, not run on hardware
//
#include <hip/hip_runtime.h>
#include <hip/hip_bf16.h>

// HSIC(X, Y) for X,Y in R^{8192 x 512}, sigma = 1.
//
//   sum(Kc * Lc) = S_KL - (2/n) * sum_i rowK_i*rowL_i + sumK*sumL / n^2
//
// The Gram matrices dominate (1.4e11 FLOPs, data fits in L2), so dot products
// run on V_WMMA_F32_16X16X32_BF16 using an error-compensated bf16 split:
//   x = hi + lo (both bf16),  <x,y> ~= hi*hi + hi*lo + lo*hi  (f32 accum)
// -> ~f32 accuracy at the bf16 matrix-pipe rate (8x the MACs/instruction of
// the f32 16x16x4 shape, 3 products instead of 8 chunks => ~2.7x fewer WMMAs).

typedef __attribute__((ext_vector_type(8)))  float  v8f;
typedef __attribute__((ext_vector_type(4)))  __bf16 v4bf;
typedef __attribute__((ext_vector_type(8)))  __bf16 v8bf;
typedef __attribute__((ext_vector_type(16))) __bf16 v16bf;

#define HSIC_N 8192
#define HSIC_D 512

static __device__ inline v8f wmma_bf16(v16bf a, v16bf b, v8f c) {
  return __builtin_amdgcn_wmma_f32_16x16x32_bf16(false, a, false, b,
                                                 (short)0, c, false, false);
}

// A fragment (16x32 bf16): lane holds one 16-half run at K = k + 16*half.
static __device__ inline v16bf load_afrag(const __bf16* rowbase, int k, int half) {
  return *(const v16bf*)(rowbase + k + 16 * half);
}

// B fragment (32x16 bf16): lane holds one 16-half run at K = k + 16*half.
static __device__ inline v16bf load_bfrag(const __bf16* rowbase, int k, int half) {
  return *(const v16bf*)(rowbase + k + 16 * half);
}

// ---------------------------------------------------------------------------
// Kernel 0: split f32 -> bf16 hi/lo pair (row-major, same shape).
//   blockIdx.y: 0 -> X, 1 -> Y. Each thread converts one float4.
// ---------------------------------------------------------------------------
__global__ __launch_bounds__(256) void hsic_split(
    const float* __restrict__ X, const float* __restrict__ Y,
    __bf16* __restrict__ Xhi, __bf16* __restrict__ Xlo,
    __bf16* __restrict__ Yhi, __bf16* __restrict__ Ylo) {
  const float* src;
  __bf16 *hi, *lo;
  if (blockIdx.y == 0) { src = X; hi = Xhi; lo = Xlo; }
  else                 { src = Y; hi = Yhi; lo = Ylo; }

  const size_t idx4 = (size_t)blockIdx.x * 256 + threadIdx.x;  // float4 index
  float4 q = ((const float4*)src)[idx4];

  v4bf hv, lv;
  float e[4] = {q.x, q.y, q.z, q.w};
#pragma unroll
  for (int c = 0; c < 4; ++c) {
    __bf16 h = (__bf16)e[c];            // round-to-nearest-even
    __bf16 l = (__bf16)(e[c] - (float)h);
    hv[c] = h;
    lv[c] = l;
  }
  *(v4bf*)(hi + idx4 * 4) = hv;
  *(v4bf*)(lo + idx4 * 4) = lv;
}

// ---------------------------------------------------------------------------
// Kernel 1: per-row squared norms in f32 (exact). One wave per row.
// ---------------------------------------------------------------------------
__global__ __launch_bounds__(256) void hsic_row_sqnorms(
    const float* __restrict__ X, const float* __restrict__ Y,
    float* __restrict__ sqX, float* __restrict__ sqY) {
  const int gw   = (blockIdx.x * 256 + threadIdx.x) >> 5;
  const int lane = threadIdx.x & 31;

  const float* src;
  float*       dst;
  int          row;
  if (gw < HSIC_N) { src = X; dst = sqX; row = gw; }
  else             { src = Y; dst = sqY; row = gw - HSIC_N; }

  const float4* p = (const float4*)(src + (size_t)row * HSIC_D);
  float s = 0.0f;
#pragma unroll
  for (int t = 0; t < 4; ++t) {
    float4 q = p[lane + 32 * t];
    s += q.x * q.x + q.y * q.y + q.z * q.z + q.w * q.w;
  }
#pragma unroll
  for (int m = 16; m >= 1; m >>= 1) s += __shfl_xor(s, m);
  if (lane == 0) dst[row] = s;
}

// ---------------------------------------------------------------------------
// Kernel 2: main tile sweep.
//   grid = (128, 2): blockIdx.x -> 64-row i-band, blockIdx.y -> j half.
//   8 waves = 2 (i-sub, 32 rows) x 4 (j-phase). Wave micro-tile: 32x32 =
//   2x2 WMMA tiles per Gram matrix; X and Y k-sweeps run back-to-back per
//   j-patch to reuse fragment registers (keeps VGPR pressure in check).
// ---------------------------------------------------------------------------
__global__ __launch_bounds__(256) void hsic_tiles(
    const __bf16* __restrict__ Xhi, const __bf16* __restrict__ Xlo,
    const __bf16* __restrict__ Yhi, const __bf16* __restrict__ Ylo,
    const float* __restrict__ sqX, const float* __restrict__ sqY,
    float* __restrict__ rowK, float* __restrict__ rowL,
    float* __restrict__ sklPart) {
  const int lane = threadIdx.x & 31;
  const int wave = threadIdx.x >> 5;
  const int wi   = wave >> 2;         // 0..1 : i sub-band (32 rows)
  const int wj   = wave & 3;          // 0..3 : j phase
  const int half = lane >> 4;
  const int l16  = lane & 15;

  const int i0    = blockIdx.x * 64 + wi * 32;
  const int jbase = blockIdx.y * (HSIC_N / 2);

  // A-side row bases (j-invariant).
  const __bf16* A0hx = Xhi + (size_t)(i0 + l16) * HSIC_D;
  const __bf16* A1hx = A0hx + (size_t)16 * HSIC_D;
  const __bf16* A0lx = Xlo + (size_t)(i0 + l16) * HSIC_D;
  const __bf16* A1lx = A0lx + (size_t)16 * HSIC_D;
  const __bf16* A0hy = Yhi + (size_t)(i0 + l16) * HSIC_D;
  const __bf16* A1hy = A0hy + (size_t)16 * HSIC_D;
  const __bf16* A0ly = Ylo + (size_t)(i0 + l16) * HSIC_D;
  const __bf16* A1ly = A0ly + (size_t)16 * HSIC_D;

  // Per-lane M-row squared norms: i-tile t, M = i0 + 16*t + v + 8*half.
  float sqAx[2][8], sqAy[2][8];
#pragma unroll
  for (int t = 0; t < 2; ++t)
#pragma unroll
    for (int v = 0; v < 8; ++v) {
      const int m = i0 + 16 * t + v + 8 * half;
      sqAx[t][v] = sqX[m];
      sqAy[t][v] = sqY[m];
    }

  float sKL = 0.0f;
  float rK[2][8], rL[2][8];
#pragma unroll
  for (int t = 0; t < 2; ++t)
#pragma unroll
    for (int v = 0; v < 8; ++v) { rK[t][v] = 0.0f; rL[t][v] = 0.0f; }

  for (int j0 = jbase + wj * 32; j0 < jbase + HSIC_N / 2; j0 += 128) {
    v8f aX00 = {}, aX01 = {}, aX10 = {}, aX11 = {};
    v8f aY00 = {}, aY01 = {}, aY10 = {}, aY11 = {};

    {  // ----- X Gram pass -----
      const __bf16* B0h = Xhi + (size_t)(j0 + l16) * HSIC_D;
      const __bf16* B1h = B0h + (size_t)16 * HSIC_D;
      const __bf16* B0l = Xlo + (size_t)(j0 + l16) * HSIC_D;
      const __bf16* B1l = B0l + (size_t)16 * HSIC_D;
#pragma unroll 2
      for (int k = 0; k < HSIC_D; k += 32) {
        v16bf a0h = load_afrag(A0hx, k, half);
        v16bf a1h = load_afrag(A1hx, k, half);
        v16bf a0l = load_afrag(A0lx, k, half);
        v16bf a1l = load_afrag(A1lx, k, half);
        v16bf b0h = load_bfrag(B0h, k, half);
        v16bf b1h = load_bfrag(B1h, k, half);
        v16bf b0l = load_bfrag(B0l, k, half);
        v16bf b1l = load_bfrag(B1l, k, half);
        aX00 = wmma_bf16(a0h, b0h, aX00);
        aX01 = wmma_bf16(a0h, b1h, aX01);
        aX10 = wmma_bf16(a1h, b0h, aX10);
        aX11 = wmma_bf16(a1h, b1h, aX11);
        aX00 = wmma_bf16(a0h, b0l, aX00);
        aX01 = wmma_bf16(a0h, b1l, aX01);
        aX10 = wmma_bf16(a1h, b0l, aX10);
        aX11 = wmma_bf16(a1h, b1l, aX11);
        aX00 = wmma_bf16(a0l, b0h, aX00);
        aX01 = wmma_bf16(a0l, b1h, aX01);
        aX10 = wmma_bf16(a1l, b0h, aX10);
        aX11 = wmma_bf16(a1l, b1h, aX11);
      }
    }
    {  // ----- Y Gram pass -----
      const __bf16* B0h = Yhi + (size_t)(j0 + l16) * HSIC_D;
      const __bf16* B1h = B0h + (size_t)16 * HSIC_D;
      const __bf16* B0l = Ylo + (size_t)(j0 + l16) * HSIC_D;
      const __bf16* B1l = B0l + (size_t)16 * HSIC_D;
#pragma unroll 2
      for (int k = 0; k < HSIC_D; k += 32) {
        v16bf a0h = load_afrag(A0hy, k, half);
        v16bf a1h = load_afrag(A1hy, k, half);
        v16bf a0l = load_afrag(A0ly, k, half);
        v16bf a1l = load_afrag(A1ly, k, half);
        v16bf b0h = load_bfrag(B0h, k, half);
        v16bf b1h = load_bfrag(B1h, k, half);
        v16bf b0l = load_bfrag(B0l, k, half);
        v16bf b1l = load_bfrag(B1l, k, half);
        aY00 = wmma_bf16(a0h, b0h, aY00);
        aY01 = wmma_bf16(a0h, b1h, aY01);
        aY10 = wmma_bf16(a1h, b0h, aY10);
        aY11 = wmma_bf16(a1h, b1h, aY11);
        aY00 = wmma_bf16(a0h, b0l, aY00);
        aY01 = wmma_bf16(a0h, b1l, aY01);
        aY10 = wmma_bf16(a1h, b0l, aY10);
        aY11 = wmma_bf16(a1h, b1l, aY11);
        aY00 = wmma_bf16(a0l, b0h, aY00);
        aY01 = wmma_bf16(a0l, b1h, aY01);
        aY10 = wmma_bf16(a1l, b0h, aY10);
        aY11 = wmma_bf16(a1l, b1h, aY11);
      }
    }

    // Epilogue: Gram -> RBF values, accumulate S_KL and row sums.
    const float sqBx0 = sqX[j0 + l16];
    const float sqBx1 = sqX[j0 + 16 + l16];
    const float sqBy0 = sqY[j0 + l16];
    const float sqBy1 = sqY[j0 + 16 + l16];
#pragma unroll
    for (int v = 0; v < 8; ++v) {
      float k00 = __expf(-0.5f * fmaxf(sqAx[0][v] + sqBx0 - 2.0f * aX00[v], 0.0f));
      float l00 = __expf(-0.5f * fmaxf(sqAy[0][v] + sqBy0 - 2.0f * aY00[v], 0.0f));
      float k01 = __expf(-0.5f * fmaxf(sqAx[0][v] + sqBx1 - 2.0f * aX01[v], 0.0f));
      float l01 = __expf(-0.5f * fmaxf(sqAy[0][v] + sqBy1 - 2.0f * aY01[v], 0.0f));
      float k10 = __expf(-0.5f * fmaxf(sqAx[1][v] + sqBx0 - 2.0f * aX10[v], 0.0f));
      float l10 = __expf(-0.5f * fmaxf(sqAy[1][v] + sqBy0 - 2.0f * aY10[v], 0.0f));
      float k11 = __expf(-0.5f * fmaxf(sqAx[1][v] + sqBx1 - 2.0f * aX11[v], 0.0f));
      float l11 = __expf(-0.5f * fmaxf(sqAy[1][v] + sqBy1 - 2.0f * aY11[v], 0.0f));

      sKL += k00 * l00 + k01 * l01 + k10 * l10 + k11 * l11;
      rK[0][v] += k00 + k01;
      rK[1][v] += k10 + k11;
      rL[0][v] += l00 + l01;
      rL[1][v] += l10 + l11;
    }
  }

  // Row sums: reduce across the 16 lanes of each half (masks 1..8 never flip
  // lane bit 4). Fixed order -> deterministic.
#pragma unroll
  for (int m = 8; m >= 1; m >>= 1)
#pragma unroll
    for (int t = 0; t < 2; ++t)
#pragma unroll
      for (int v = 0; v < 8; ++v) {
        rK[t][v] += __shfl_xor(rK[t][v], m);
        rL[t][v] += __shfl_xor(rL[t][v], m);
      }
#pragma unroll
  for (int m = 16; m >= 1; m >>= 1) sKL += __shfl_xor(sKL, m);

  __shared__ float sRowK[2][4][32];
  __shared__ float sRowL[2][4][32];
  __shared__ float sSkl[8];
  if (l16 == 0) {
#pragma unroll
    for (int t = 0; t < 2; ++t)
#pragma unroll
      for (int v = 0; v < 8; ++v) {
        sRowK[wi][wj][16 * t + 8 * half + v] = rK[t][v];
        sRowL[wi][wj][16 * t + 8 * half + v] = rL[t][v];
      }
  }
  if (lane == 0) sSkl[wave] = sKL;
  __syncthreads();

  const int tid = threadIdx.x;
  if (tid < 64) {
    const int rwi = tid >> 5, rm = tid & 31;
    float k = sRowK[rwi][0][rm] + sRowK[rwi][1][rm] +
              sRowK[rwi][2][rm] + sRowK[rwi][3][rm];
    float l = sRowL[rwi][0][rm] + sRowL[rwi][1][rm] +
              sRowL[rwi][2][rm] + sRowL[rwi][3][rm];
    const int row = blockIdx.x * 64 + tid;
    rowK[blockIdx.y * HSIC_N + row] = k;
    rowL[blockIdx.y * HSIC_N + row] = l;
  }
  if (tid == 0) {
    float s = 0.0f;
#pragma unroll
    for (int w = 0; w < 8; ++w) s += sSkl[w];
    sklPart[blockIdx.y * 128 + blockIdx.x] = s;
  }
}

// ---------------------------------------------------------------------------
// Kernel 3: deterministic final reduction + HSIC formula.
// ---------------------------------------------------------------------------
__global__ __launch_bounds__(256) void hsic_finalize(
    const float* __restrict__ sklPart, const float* __restrict__ rowK,
    const float* __restrict__ rowL, float* __restrict__ out) {
  __shared__ float sK[256], sL[256], sD[256], sS[256];
  const int t = threadIdx.x;

  float aK = 0.0f, aL = 0.0f, aD = 0.0f;
  for (int i = t; i < HSIC_N; i += 256) {
    float k = rowK[i] + rowK[HSIC_N + i];
    float l = rowL[i] + rowL[HSIC_N + i];
    aK += k; aL += l; aD += k * l;
  }
  sK[t] = aK; sL[t] = aL; sD[t] = aD; sS[t] = sklPart[t];
  __syncthreads();
  for (int off = 128; off > 0; off >>= 1) {
    if (t < off) {
      sK[t] += sK[t + off];
      sL[t] += sL[t + off];
      sD[t] += sD[t + off];
      sS[t] += sS[t + off];
    }
    __syncthreads();
  }
  if (t == 0) {
    const float n = (float)HSIC_N;
    float hsic = (sS[0] - (2.0f / n) * sD[0] + (sK[0] * sL[0]) / (n * n)) /
                 ((n - 1.0f) * (n - 1.0f));
    out[0] = hsic;
  }
}

extern "C" void kernel_launch(void* const* d_in, const int* in_sizes, int n_in,
                              void* d_out, int out_size, void* d_ws, size_t ws_size,
                              hipStream_t stream) {
  (void)in_sizes; (void)n_in; (void)out_size; (void)ws_size;
  const float* X = (const float*)d_in[0];
  const float* Y = (const float*)d_in[1];
  float* out = (float*)d_out;

  // Workspace layout. All scratch is written before read within this launch
  // sequence -> replay-safe without zero-init.
  float* ws      = (float*)d_ws;
  float* sklPart = ws;                    // [256]
  float* rowK    = ws + 256;              // [2*8192]
  float* rowL    = rowK + 2 * HSIC_N;     // [2*8192]
  float* sqX     = rowL + 2 * HSIC_N;     // [8192]
  float* sqY     = sqX + HSIC_N;          // [8192]
  // bf16 split arrays (offset 49408 floats -> 32B aligned).
  __bf16* Xhi = (__bf16*)(sqY + HSIC_N);
  __bf16* Xlo = Xhi + (size_t)HSIC_N * HSIC_D;
  __bf16* Yhi = Xlo + (size_t)HSIC_N * HSIC_D;
  __bf16* Ylo = Yhi + (size_t)HSIC_N * HSIC_D;

  // 0) f32 -> bf16 hi/lo split: 1M float4s per matrix -> grid (4096, 2).
  hipLaunchKernelGGL(hsic_split, dim3(4096, 2), dim3(256), 0, stream,
                     X, Y, Xhi, Xlo, Yhi, Ylo);
  // 1) exact f32 row squared norms.
  hipLaunchKernelGGL(hsic_row_sqnorms, dim3(2048), dim3(256), 0, stream,
                     X, Y, sqX, sqY);
  // 2) main WMMA tile sweep: (128 i-bands) x (2 j-halves) x 256 threads.
  hipLaunchKernelGGL(hsic_tiles, dim3(128, 2), dim3(256), 0, stream,
                     Xhi, Xlo, Yhi, Ylo, sqX, sqY, rowK, rowL, sklPart);
  // 3) finalize.
  hipLaunchKernelGGL(hsic_finalize, dim3(1), dim3(256), 0, stream,
                     sklPart, rowK, rowL, out);
}
